// KernelPooling_80049600463687
// MI455X (gfx1250) — matrix operand order repaired and checked by reading, not compiled, not run on hardware
//
#include <hip/hip_runtime.h>
#include <hip/hip_bf16.h>
#include <math.h>

typedef __attribute__((ext_vector_type(16))) _Float16 v16h;
typedef __attribute__((ext_vector_type(8)))  _Float16 v8h;
typedef __attribute__((ext_vector_type(8)))  float    v8f;

#define C_DIM 512
#define D_DIM 4096
#define HW_DIM 784
#define B_DIM 8
#define NLOC (B_DIM * HW_DIM)           // 6272
#define TILES_PER_B (HW_DIM / 16)       // 49
#define LDS_STRIDE 520                  // halves; 1040B = 65*16 -> 16B aligned rows, bank stride 4
#define TWO_PI_OVER_N 1.5339807878856412e-3f  // 2*pi/4096

// ---------------- zero scratch accumulators ----------------
__global__ void tsk_zero(float* __restrict__ p, int nelem) {
    int i = blockIdx.x * 256 + threadIdx.x;
    if (i < nelem) p[i] = 0.0f;
}

// ---------------- build sketch-DFT matrix E (f16) ----------------
// E[(t*2+reim)*D + n][c] = s[t,c] * {cos,sin}(-2*pi*h[t,c]*n/4096)
__global__ void tsk_build_e(const int* __restrict__ h_idx,
                            const int* __restrict__ s_bits,
                            _Float16* __restrict__ E) {
    int gid = blockIdx.x * 256 + threadIdx.x;      // 3*4096*512 total
    if (gid >= 3 * D_DIM * C_DIM) return;
    int c = gid & (C_DIM - 1);
    int n = (gid >> 9) & (D_DIM - 1);
    int t = gid >> 21;
    int h = h_idx[t * C_DIM + c];
    float s = (float)(2 * s_bits[t * C_DIM + c] - 1);
    int ph = (h * n) & (D_DIM - 1);
    float th = -(float)ph * TWO_PI_OVER_N;         // forward FFT: e^{-i theta}
    float sv, cv;
    __sincosf(th, &sv, &cv);
    E[((size_t)(t * 2 + 0) * D_DIM + n) * C_DIM + c] = (_Float16)(s * cv);
    E[((size_t)(t * 2 + 1) * D_DIM + n) * C_DIM + c] = (_Float16)(s * sv);
}

// ---------------- fused sketch-DFT GEMM + order cumprod + spatial reduce ----
// grid: (D/128, NLOC/16), block 256 (8 waves). Wave computes 16x16 outputs of
// 6 real GEMMs (3 orders x re/im) over K=512 via v_wmma_f32_16x16x32_f16.
__global__ __launch_bounds__(256) void tsk_gemm(const float* __restrict__ x,
                                                const _Float16* __restrict__ E,
                                                float* __restrict__ acc) {
    __shared__ _Float16 ldsA[16 * LDS_STRIDE];
    const int tid   = threadIdx.x;
    const int tileN = blockIdx.x;                  // 0..31
    const int tileM = blockIdx.y;                  // 0..391
    const int b     = tileM / TILES_PER_B;         // whole tile in one batch (784%16==0)

    // stage A tile: 16 locations x 512 channels, fp32 -> f16
    const float* xrow = x + (size_t)tileM * 16 * C_DIM;
    for (int i = tid; i < 16 * C_DIM; i += 256) {
        int r = i >> 9, c = i & (C_DIM - 1);
        ldsA[r * LDS_STRIDE + c] = (_Float16)xrow[i];
    }
    __syncthreads();

    const int wave  = tid >> 5;
    const int lane  = tid & 31;
    const int lm    = lane & 15;
    const int khalf = lane >> 4;
    const int n     = tileN * 128 + wave * 16 + lm;

    const _Float16* bptr[6];
#pragma unroll
    for (int q = 0; q < 6; ++q)                    // q = t*2 + reim
        bptr[q] = E + ((size_t)q * D_DIM + n) * C_DIM;

    v8f accm[6] = {};
#pragma unroll
    for (int kt = 0; kt < 16; ++kt) {              // K = 512 = 16 * 32
        const int cbase = kt * 32;
        // A fragment (contiguous k order, same as B): lane half selects K 0-15 vs 16-31
        const _Float16* ap = &ldsA[lm * LDS_STRIDE + cbase + khalf * 16];
        v8h alo = *(const v8h*)ap;
        v8h ahi = *(const v8h*)(ap + 8);
        v16h a = __builtin_shufflevector(alo, ahi, 0,1,2,3,4,5,6,7,8,9,10,11,12,13,14,15);
#pragma unroll
        for (int q = 0; q < 6; ++q) {
            // B fragment (16-bit B 32x16): lane = column n, 16 contiguous K halves
            v16h bb = *(const v16h*)(bptr[q] + cbase + khalf * 16);
            accm[q] = __builtin_amdgcn_wmma_f32_16x16x32_f16(
                false, a, false, bb, (short)0, accm[q], false, false);
        }
    }

    // cumprod over orders per (m,k) element, then reduce over the 16 locations.
    // C/D layout: VGPR v holds row M = v + 8*khalf, column N = lane&15.
    float s1r = 0.f, s1i = 0.f, s2r = 0.f, s2i = 0.f;
#pragma unroll
    for (int v = 0; v < 8; ++v) {
        float r0 = accm[0][v], i0 = accm[1][v];
        float r1 = accm[2][v], i1 = accm[3][v];
        float r2 = accm[4][v], i2 = accm[5][v];
        float c1r = r0 * r1 - i0 * i1;             // xf0*xf1
        float c1i = r0 * i1 + i0 * r1;
        s1r += c1r; s1i += c1i;
        s2r += c1r * r2 - c1i * i2;                // xf0*xf1*xf2
        s2i += c1r * i2 + c1i * r2;
    }
    s1r += __shfl_xor(s1r, 16, 32);                // combine rows 0-7 with 8-15
    s1i += __shfl_xor(s1i, 16, 32);
    s2r += __shfl_xor(s2r, 16, 32);
    s2i += __shfl_xor(s2i, 16, 32);
    if (lane < 16) {
        atomicAdd(acc + ((size_t)(b * 2 + 0) * 2 + 0) * D_DIM + n, s1r);
        atomicAdd(acc + ((size_t)(b * 2 + 0) * 2 + 1) * D_DIM + n, s1i);
        atomicAdd(acc + ((size_t)(b * 2 + 1) * 2 + 0) * D_DIM + n, s2r);
        atomicAdd(acc + ((size_t)(b * 2 + 1) * 2 + 1) * D_DIM + n, s2i);
    }
}

// ---------------- per-(batch,channel) spatial sum -------------------------
__global__ void tsk_chansum(const float* __restrict__ x, float* __restrict__ cs) {
    int b = blockIdx.x;
    int c = threadIdx.x;                            // 512 threads
    const float* p = x + (size_t)b * HW_DIM * C_DIM + c;
    float s = 0.f;
    for (int l = 0; l < HW_DIM; ++l) s += p[(size_t)l * C_DIM];
    cs[b * C_DIM + c] = s;
}

// ---------------- inverse DFT (real part) of accumulated products ---------
// xi[b,ord,n] = (1/(4096*784)) * sum_k [Re cos(2pi k n/N) - Im sin(2pi k n/N)]
__global__ void tsk_ifft(const float* __restrict__ acc, float* __restrict__ xi) {
    int gid = blockIdx.x * 256 + threadIdx.x;       // 8*2*4096 = 65536
    int n  = gid & (D_DIM - 1);
    int bo = gid >> 12;                             // b*2+ord
    const float* re = acc + ((size_t)bo * 2 + 0) * D_DIM;
    const float* im = acc + ((size_t)bo * 2 + 1) * D_DIM;
    float s = 0.f;
    for (int k = 0; k < D_DIM; ++k) {
        int ph = (k * n) & (D_DIM - 1);
        float th = (float)ph * TWO_PI_OVER_N;       // inverse: e^{+i theta}
        float sv, cv;
        __sincosf(th, &sv, &cv);
        s += re[k] * cv - im[k] * sv;
    }
    xi[gid] = s * (1.0f / (4096.0f * 784.0f));
}

// ---------------- assemble phi, signed sqrt, L2 normalize -----------------
__global__ __launch_bounds__(256) void tsk_finalize(const float* __restrict__ alpha,
                                                    const float* __restrict__ cs,
                                                    const float* __restrict__ xi,
                                                    float* __restrict__ out) {
    __shared__ float red[256];
    const int b = blockIdx.x;
    const int tid = threadIdx.x;
    const int PHI = 1 + C_DIM + 2 * D_DIM;          // 8705
    float vals[35];
    float ssum = 0.f;
    int cnt = 0;
    float a0 = alpha[0], a1 = alpha[1], a2 = alpha[2], a3 = alpha[3];
    for (int i = tid; i < PHI; i += 256) {
        float v;
        if (i == 0)        v = a0;
        else if (i <= 512) v = a1 * cs[b * C_DIM + (i - 1)] * (1.0f / 784.0f);
        else {
            int j = i - 513;
            int ord = j >> 12;
            int n = j & (D_DIM - 1);
            v = (ord ? a3 : a2) * xi[((size_t)(b * 2 + ord)) * D_DIM + n];
        }
        vals[cnt++] = v;
        ssum += fabsf(v) + 1e-12f;                  // == |signed_sqrt(v)|^2
    }
    red[tid] = ssum;
    __syncthreads();
    for (int ofs = 128; ofs > 0; ofs >>= 1) {
        if (tid < ofs) red[tid] += red[tid + ofs];
        __syncthreads();
    }
    float inv = 1.0f / sqrtf(red[0]);
    cnt = 0;
    for (int i = tid; i < PHI; i += 256) {
        float v = vals[cnt++];
        float sg = (v > 0.f) ? 1.f : ((v < 0.f) ? -1.f : 0.f);
        out[(size_t)b * PHI + i] = sg * sqrtf(fabsf(v) + 1e-12f) * inv;
    }
}

extern "C" void kernel_launch(void* const* d_in, const int* in_sizes, int n_in,
                              void* d_out, int out_size, void* d_ws, size_t ws_size,
                              hipStream_t stream) {
    (void)in_sizes; (void)n_in; (void)out_size; (void)ws_size;
    const float* x      = (const float*)d_in[0];   // (8,28,28,512) f32
    const float* alpha  = (const float*)d_in[1];   // (4,) f32
    const int*   h_idx  = (const int*)d_in[2];     // (3,512) i32
    const int*   s_bits = (const int*)d_in[3];     // (3,512) i32
    float* out = (float*)d_out;

    char* ws = (char*)d_ws;
    float*    acc = (float*)(ws + 0);              // 8*2*2*4096 f32 = 512 KB
    float*    xi  = (float*)(ws + 524288);         // 8*2*4096   f32 = 256 KB
    float*    cs  = (float*)(ws + 786432);         // 8*512      f32 =  16 KB
    _Float16* E   = (_Float16*)(ws + 1048576);     // 3*2*4096*512 f16 = 24 MB

    tsk_zero   <<<512, 256, 0, stream>>>(acc, B_DIM * 2 * 2 * D_DIM);
    tsk_build_e<<<(3 * D_DIM * C_DIM) / 256, 256, 0, stream>>>(h_idx, s_bits, E);
    tsk_gemm   <<<dim3(D_DIM / 128, NLOC / 16), 256, 0, stream>>>(x, E, acc);
    tsk_chansum<<<B_DIM, C_DIM, 0, stream>>>(x, cs);
    tsk_ifft   <<<(B_DIM * 2 * D_DIM) / 256, 256, 0, stream>>>(acc, xi);
    tsk_finalize<<<B_DIM, 256, 0, stream>>>(alpha, cs, xi, out);
}
